// SegmentationDilation_52871047413903
// MI455X (gfx1250) — 8-bit matrix operand order repaired and checked by reading, compiled, not run on hardware
//
#include <hip/hip_runtime.h>
#include <hip/hip_bf16.h>
#include <math.h>


typedef __attribute__((ext_vector_type(8))) int v8i;
typedef int v4i_g __attribute__((vector_size(4 * sizeof(int))));
typedef __attribute__((address_space(1))) v4i_g* gl_v4i_ptr;
typedef __attribute__((address_space(3))) v4i_g* ds_v4i_ptr;

#define IMGS   8
#define HH     512
#define WW     512
#define WPR    16
#define WORDS  (HH * WPR)
#define RAD    10
#define PASSES 51

#if defined(__gfx1250__) && __has_builtin(__builtin_amdgcn_global_load_async_to_lds_b128)
#define HAVE_ASYNC_LDS 1
#else
#define HAVE_ASYNC_LDS 0
#endif

__global__ void seg_detseed_kernel(const float2* __restrict__ in,
                                   unsigned* __restrict__ detg,
                                   unsigned* __restrict__ seeds) {
    const int img  = blockIdx.x;
    const int lane = threadIdx.x & 31;
    const int wv   = threadIdx.x >> 5;
    const float2* ip = in + (size_t)img * (HH * WW);

    float    bd = -INFINITY;
    unsigned bi = 0u;

    for (int j = 0; j < 512; ++j) {
        const int wordIdx = wv * 512 + j;
        const int row = wordIdx >> 4;
        const int wc  = wordIdx & 15;
        const int col = (wc << 5) + lane;
        const float2 v = ip[row * WW + col];
        const float d = v.y - v.x;
        const unsigned flat = (unsigned)(row * WW + col);
        if (d > bd) { bd = d; bi = flat; }
        const unsigned long long bal = __ballot(d > 0.0f);
        if (lane == 0) detg[img * WORDS + wordIdx] = (unsigned)bal;
    }

    for (int off = 16; off > 0; off >>= 1) {
        const float    od = __shfl_down(bd, off, 32);
        const unsigned oi = __shfl_down(bi, off, 32);
        if (od > bd || (od == bd && oi < bi)) { bd = od; bi = oi; }
    }
    __shared__ float    sd[16];
    __shared__ unsigned si[16];
    if (lane == 0) { sd[wv] = bd; si[wv] = bi; }
    __syncthreads();
    if (threadIdx.x == 0) {
        float    best = sd[0];
        unsigned bidx = si[0];
        for (int w = 1; w < 16; ++w)
            if (sd[w] > best || (sd[w] == best && si[w] < bidx)) { best = sd[w]; bidx = si[w]; }
        seeds[img] = bidx;
    }
}

__global__ void seg_grow_kernel(const unsigned* __restrict__ seeds,
                                unsigned* __restrict__ detg) {
    extern __shared__ unsigned smem[];
    unsigned* det_s = smem;
    unsigned* obj_s = smem + WORDS;
    unsigned* tmp_s = smem + 2 * WORDS;

    const int img  = blockIdx.x;
    const int tid  = threadIdx.x;
    const int lane = tid & 31;
    const int wv   = tid >> 5;
    unsigned* dg = detg + img * WORDS;

#if HAVE_ASYNC_LDS
    for (int k = 0; k < 2; ++k) {
        const int j = (tid + k * 1024) * 4;
        __builtin_amdgcn_global_load_async_to_lds_b128(
            (gl_v4i_ptr)(dg + j),
            (ds_v4i_ptr)(det_s + j),
            0, 0);
    }
    for (int k = 0; k < 8; ++k) obj_s[tid + k * 1024] = 0u;
#if __has_builtin(__builtin_amdgcn_s_wait_asynccnt)
    __builtin_amdgcn_s_wait_asynccnt(0);
#else
    asm volatile("s_wait_asynccnt 0" ::: "memory");
#endif
#else
    for (int k = 0; k < 8; ++k) {
        const int i = tid + k * 1024;
        det_s[i] = dg[i];
        obj_s[i] = 0u;
    }
#endif
    __syncthreads();
    if (tid == 0) {
        const unsigned f   = seeds[img];
        const unsigned row = f >> 9;
        const unsigned col = f & 511u;
        obj_s[row * WPR + (col >> 5)] = 1u << (col & 31);
    }

    v8i amat;
    {
        const int m    = lane & 15;
        const int half = lane >> 4;
#pragma unroll
        for (int v = 0; v < 8; ++v) {
            unsigned wA = 0u;
#pragma unroll
            for (int bb = 0; bb < 4; ++bb) {
                const int k = ((v >> 2) << 5) + (((v >> 1) & 1) << 4) + (half << 3) + ((v & 1) << 2) + bb;
                const unsigned one = (k >= m && k <= m + 20) ? 1u : 0u;
                wA |= one << (bb << 3);
            }
            amat[v] = (int)wA;
        }
    }

    const int half = lane >> 4;
    const int nidx = lane & 15;

    for (int pass = 0; pass < PASSES; ++pass) {
        __syncthreads();

        for (int k = 0; k < 8; ++k) {
            const int i  = tid + k * 1024;
            const int wc = i & 15;
            const unsigned C = obj_s[i];
            const unsigned L = wc ? obj_s[i - 1] : 0u;
            const unsigned R = (wc < 15) ? obj_s[i + 1] : 0u;
            unsigned long long Y = (((unsigned long long)C << 32) | L) >> 22;
            Y |= Y >> 1; Y |= Y >> 2; Y |= Y >> 4; Y |= Y >> 2;
            unsigned long long Z = ((((unsigned long long)R << 32) | C) >> 1);
            Z |= Z >> 1; Z |= Z >> 2; Z |= Z >> 4; Z |= Z >> 2;
            tmp_s[i] = C | (unsigned)Y | (unsigned)Z;
        }
        __syncthreads();

        for (int q = 0; q < 16; ++q) {
            const int pairIdx = wv * 16 + q;
            const int rowtile = pairIdx >> 4;
            const int wcpair  = pairIdx & 15;
            const int r0 = rowtile << 4;

            unsigned balA[8], balB[8];
#pragma unroll
            for (int t = 0; t < 2; ++t) {
                const int col   = (wcpair << 5) + (t << 4) + nidx;
                const int shift = col & 31;
                v8i bmat;
#pragma unroll
                for (int v = 0; v < 8; ++v) {
                    unsigned wB = 0u;
                    const int kb = ((v >> 2) << 5) + (((v >> 1) & 1) << 4) + (half << 3) + ((v & 1) << 2);
#pragma unroll
                    for (int bb = 0; bb < 4; ++bb) {
                        const int row = r0 - RAD + kb + bb;
                        const unsigned inb = ((unsigned)row < (unsigned)HH) ? 1u : 0u;
                        const unsigned bit = (tmp_s[((row & 511) << 4) + wcpair] >> shift) & 1u;
                        wB |= (bit & inb) << (bb << 3);
                    }
                    bmat[v] = (int)wB;
                }
                v8i cacc = {0, 0, 0, 0, 0, 0, 0, 0};
                v8i dres = __builtin_amdgcn_wmma_i32_16x16x64_iu8(
                    false, amat, false, bmat, cacc, false, false);
#pragma unroll
                for (int j = 0; j < 8; ++j) {
                    const unsigned bb32 = (unsigned)__ballot(dres[j] > 0);
                    if (t == 0) balA[j] = bb32; else balB[j] = bb32;
                }
            }
#pragma unroll
            for (int j = 0; j < 8; ++j) {
                const unsigned lowRow  = (balA[j] & 0xFFFFu) | ((balB[j] & 0xFFFFu) << 16);
                const unsigned highRow = (balA[j] >> 16)     | (balB[j] & 0xFFFF0000u);
                const int wlow  = ((r0 + j)     << 4) + wcpair;
                const int whigh = ((r0 + 8 + j) << 4) + wcpair;
                if (lane == j)     obj_s[wlow]  = lowRow  & det_s[wlow];
                if (lane == 8 + j) obj_s[whigh] = highRow & det_s[whigh];
            }
        }
    }
    __syncthreads();
    for (int k = 0; k < 8; ++k) {
        const int i = tid + k * 1024;
        dg[i] = obj_s[i];
    }
}

__global__ void seg_expand_kernel(const unsigned* __restrict__ objg,
                                  float4* __restrict__ out) {
    const unsigned tid = blockIdx.x * 256u + threadIdx.x;
    const unsigned o4  = tid << 2;
    if (o4 >= 64u * HH * WW) return;
    const unsigned img = (o4 >> 18) & 7u;
    const unsigned pix = o4 & 262143u;
    const unsigned row = pix >> 9;
    const unsigned cb  = pix & 511u;
    const unsigned word = objg[img * WORDS + row * WPR + (cb >> 5)];
    const unsigned sh   = cb & 31u;
    float4 r;
    r.x = ((word >> (sh + 0)) & 1u) ? 1.0f : 0.0f;
    r.y = ((word >> (sh + 1)) & 1u) ? 1.0f : 0.0f;
    r.z = ((word >> (sh + 2)) & 1u) ? 1.0f : 0.0f;
    r.w = ((word >> (sh + 3)) & 1u) ? 1.0f : 0.0f;
    out[tid] = r;
}

extern "C" void kernel_launch(void* const* d_in, const int* in_sizes, int n_in,
                              void* d_out, int out_size, void* d_ws, size_t ws_size,
                              hipStream_t stream) {
    (void)in_sizes; (void)n_in; (void)out_size; (void)ws_size;
    const float2* in = (const float2*)d_in[0];
    unsigned* ws    = (unsigned*)d_ws;
    unsigned* detg  = ws;
    unsigned* seeds = ws + IMGS * WORDS;

    seg_detseed_kernel<<<dim3(IMGS), dim3(512), 0, stream>>>(in, detg, seeds);
    seg_grow_kernel<<<dim3(IMGS), dim3(1024), 3 * WORDS * sizeof(unsigned), stream>>>(seeds, detg);

    const unsigned n4 = (64u * HH * WW) / 4u;
    seg_expand_kernel<<<dim3(n4 / 256u), dim3(256), 0, stream>>>(detg, (float4*)d_out);
}
